// CrissCrossAttention_12446815224291
// MI455X (gfx1250) — matrix operand order repaired and checked by reading, not compiled, not run on hardware
//
#include <hip/hip_runtime.h>
#include <hip/hip_bf16.h>

typedef __attribute__((ext_vector_type(16))) __bf16    v16bf;
typedef __attribute__((ext_vector_type(8)))  float     v8f;
typedef __attribute__((ext_vector_type(4)))  unsigned int uint4v;

union frag_u { uint4v q[2]; v16bf v; };

#define B_  4
#define C_  256
#define N_  4096
#define CR_ 32

__device__ inline unsigned short f2bf(float f) {
    union { float f; unsigned int u; } c; c.f = f;
    unsigned int u = c.u + 0x7FFFu + ((c.u >> 16) & 1u);
    return (unsigned short)(u >> 16);
}
__device__ inline float bf2f(unsigned short h) {
    union { unsigned int u; float f; } c; c.u = ((unsigned int)h) << 16;
    return c.f;
}

// ---- prep: x (b,c,n) f32 -> xbt (b,n,c) bf16 ------------------------------
__global__ __launch_bounds__(256) void k_transpose_x(const float* __restrict__ x,
                                                     unsigned short* __restrict__ xbt) {
    int tid = blockIdx.x * 256 + threadIdx.x;           // b*n*c elements
    int ch = tid & (C_ - 1);
    int rest = tid >> 8;
    int i  = rest & (N_ - 1);
    int bI = rest >> 12;
    xbt[tid] = f2bf(x[((size_t)bI * C_ + ch) * N_ + i]);
}

__global__ __launch_bounds__(256) void k_convert(const float* __restrict__ s,
                                                 unsigned short* __restrict__ d, int count) {
    int i = blockIdx.x * 256 + threadIdx.x;
    if (i < count) d[i] = f2bf(s[i]);
}

// ---- fused projection GEMM + BN + ReLU (WMMA bf16) ------------------------
// out = relu(BN(W (Cout x 256) * x (256 x n))) per batch.
// transposed=1 : store (b, n, Cout)   (for Q, K)
// transposed=0 : store (b, Cout, n)   (for V)
__global__ __launch_bounds__(256, 1) void k_proj(const unsigned short* __restrict__ wb,
                                                 const unsigned short* __restrict__ xbt,
                                                 unsigned short* __restrict__ outp,
                                                 const float* __restrict__ sc,
                                                 const float* __restrict__ bi,
                                                 const float* __restrict__ me,
                                                 const float* __restrict__ va,
                                                 int Cout, int transposed, int totalTiles) {
    const int lane = threadIdx.x & 31, wv = threadIdx.x >> 5;
    int tile = blockIdx.x * 8 + wv;
    if (tile >= totalTiles) return;
    const int lane16 = lane & 15, h = lane >> 4;
    const int Mtiles = Cout >> 4;
    const int Ntiles = N_ >> 4;
    int nt = tile % Ntiles;
    int rest = tile / Ntiles;
    int mt = rest % Mtiles;
    int bI = rest / Mtiles;
    int m0 = mt * 16, n0 = nt * 16;

    v8f acc = {};
    const uint4v* wrow = (const uint4v*)(wb + (size_t)(m0 + lane16) * C_);
    const unsigned short* xrow = xbt + ((size_t)bI * N_ + n0 + lane16) * C_;
#pragma unroll
    for (int k0 = 0; k0 < C_; k0 += 32) {
        frag_u fa;
        fa.q[0] = wrow[(k0 >> 3) + h];         // K = k0+{0..7} (lo) / {8..15} (hi)
        fa.q[1] = wrow[(k0 >> 3) + 2 + h];     // K = k0+{16..23} / {24..31}
        frag_u fb; fb.q[0] = *(const uint4v*)(xrow + k0 + h * 8); fb.q[1] = *(const uint4v*)(xrow + k0 + 16 + h * 8);
        v16bf bx = fb.v;
        acc = __builtin_amdgcn_wmma_f32_16x16x32_bf16(false, fa.v, false, bx,
                                                      (short)0, acc, false, false);
    }
#pragma unroll
    for (int r = 0; r < 8; ++r) {
        int ch = m0 + r + 8 * h;
        float inv = sc[ch] / sqrtf(va[ch] + 1e-5f);
        float y = (acc[r] - me[ch]) * inv + bi[ch];
        y = fmaxf(y, 0.0f);
        int col = n0 + lane16;
        size_t idx = transposed ? ((size_t)bI * N_ + col) * Cout + ch
                                : ((size_t)bI * Cout + ch) * N_ + col;
        outp[idx] = f2bf(y);
    }
}

// ---- fused criss-cross attention (flash-style online softmax, WMMA) -------
__global__ __launch_bounds__(256, 1) void k_attn(const unsigned short* __restrict__ qb, // (b,n,32)
                                                 const unsigned short* __restrict__ kb, // (b,n,32)
                                                 const unsigned short* __restrict__ vb, // (b,c,n)
                                                 const float* __restrict__ x,           // (b,c,n)
                                                 const float* __restrict__ gamma,
                                                 float* __restrict__ out) {
    __shared__ __align__(64) unsigned short plds[8 * 512];   // 16x32 bf16 per wave
    const int lane = threadIdx.x & 31, wv = threadIdx.x >> 5;
    const int lane16 = lane & 15, h = lane >> 4;
    int tile = blockIdx.x * 8 + wv;                 // b * (n/16) tiles
    int it = tile & ((N_ >> 4) - 1);
    int bI = tile >> 8;
    int I0 = it * 16;
    unsigned short* myLds = plds + wv * 512;

    // Q A-fragment: rows I0..I0+15, K = 0..31
    const uint4v* qrow = (const uint4v*)(qb + ((size_t)bI * N_ + I0 + lane16) * CR_);
    frag_u fq;
    fq.q[0] = qrow[h];
    fq.q[1] = qrow[2 + h];

    // energy_w for row I0+lane16 (== masked diagonal of QtK)
    const unsigned short* qr = qb + ((size_t)bI * N_ + I0 + lane16) * CR_;
    const unsigned short* kr = kb + ((size_t)bI * N_ + I0 + lane16) * CR_;
    float ew = 0.0f;
#pragma unroll
    for (int d = 0; d < CR_; ++d) ew += bf2f(qr[d]) * bf2f(kr[d]);

    const float g = gamma[0];

    for (int half = 0; half < 2; ++half) {
        float m[8], l[8], ewv[8];
        v8f O[8];
        v8f zf = {};
#pragma unroll
        for (int r = 0; r < 8; ++r) {
            float e = __shfl(ew, r + 8 * h, 32);
            ewv[r] = e; m[r] = e; l[r] = 1.0f;    // w-column folded into init state
            O[r] = zf;
        }

        for (int j0 = 0; j0 < N_; j0 += 32) {
            // K B-fragments for columns j0..j0+15 and j0+16..j0+31
            const unsigned short* kbase = kb + ((size_t)bI * N_ + j0 + lane16) * CR_ + h * 8;
            frag_u fk0, fk1;
            fk0.q[0] = *(const uint4v*)(kbase); fk0.q[1] = *(const uint4v*)(kbase + 16);
            fk1.q[0] = *(const uint4v*)(kbase + 16 * CR_); fk1.q[1] = *(const uint4v*)(kbase + 16 * CR_ + 16);
            v16bf bk0 = fk0.v;
            v16bf bk1 = fk1.v;
            v8f s0 = __builtin_amdgcn_wmma_f32_16x16x32_bf16(false, fq.v, false, bk0,
                                                             (short)0, zf, false, false);
            v8f s1 = __builtin_amdgcn_wmma_f32_16x16x32_bf16(false, fq.v, false, bk1,
                                                             (short)0, zf, false, false);
            float scl[8], p0[8], p1[8];
#pragma unroll
            for (int r = 0; r < 8; ++r) {
                int rowi = I0 + r + 8 * h;
                float a = s0[r], b = s1[r];
                if (rowi == j0 + lane16)      a = -1e30f;   // mask diagonal
                if (rowi == j0 + 16 + lane16) b = -1e30f;
                float t = fmaxf(a, b);
                t = fmaxf(t, __shfl_xor(t, 1, 32));
                t = fmaxf(t, __shfl_xor(t, 2, 32));
                t = fmaxf(t, __shfl_xor(t, 4, 32));
                t = fmaxf(t, __shfl_xor(t, 8, 32));
                float mn = fmaxf(m[r], t);
                float sf = __expf(m[r] - mn);
                m[r] = mn;
                float e0 = __expf(a - mn), e1 = __expf(b - mn);
                float rs = e0 + e1;
                rs += __shfl_xor(rs, 1, 32);
                rs += __shfl_xor(rs, 2, 32);
                rs += __shfl_xor(rs, 4, 32);
                rs += __shfl_xor(rs, 8, 32);
                l[r] = l[r] * sf + rs;
                scl[r] = sf; p0[r] = e0; p1[r] = e1;
            }
            // bounce P (C-layout) through LDS into A-layout bf16 fragment
#pragma unroll
            for (int r = 0; r < 8; ++r) {
                int row = r + 8 * h;
                myLds[row * 32 + lane16]      = f2bf(p0[r]);
                myLds[row * 32 + 16 + lane16] = f2bf(p1[r]);
            }
            frag_u fp;
            const uint4v* prow = (const uint4v*)(myLds + lane16 * 32);
            fp.q[0] = prow[h];
            fp.q[1] = prow[2 + h];
            // rescale running O, then O += P * V  (8 channel chunks of 16)
#pragma unroll
            for (int cc = 0; cc < 8; ++cc)
#pragma unroll
                for (int r = 0; r < 8; ++r) O[cc][r] *= scl[r];
            const unsigned short* vbase =
                vb + ((size_t)bI * C_ + half * 128 + lane16) * N_ + j0 + h * 8;
#pragma unroll
            for (int cc = 0; cc < 8; ++cc) {
                frag_u fv; fv.q[0] = *(const uint4v*)(vbase + (size_t)cc * 16 * N_); fv.q[1] = *(const uint4v*)(vbase + (size_t)cc * 16 * N_ + 16);
                v16bf bv = fv.v;
                O[cc] = __builtin_amdgcn_wmma_f32_16x16x32_bf16(false, fp.v, false, bv,
                                                                (short)0, O[cc], false, false);
            }
        }

        // epilogue: normalize, add w-branch + residual
#pragma unroll
        for (int r = 0; r < 8; ++r) {
            float invl = 1.0f / l[r];
            float attw = __expf(ewv[r] - m[r]) * invl;
            int rowi = I0 + r + 8 * h;
#pragma unroll
            for (int cc = 0; cc < 8; ++cc) {
                int c = half * 128 + cc * 16 + lane16;
                size_t idx = ((size_t)bI * C_ + c) * N_ + rowi;
                float vv = bf2f(vb[idx]);
                out[idx] = g * (O[cc][r] * invl + attw * vv) + x[idx];
            }
        }
    }
}

extern "C" void kernel_launch(void* const* d_in, const int* in_sizes, int n_in,
                              void* d_out, int out_size, void* d_ws, size_t ws_size,
                              hipStream_t stream) {
    (void)in_sizes; (void)n_in; (void)out_size; (void)ws_size;
    const float* x    = (const float*)d_in[0];
    const float* w_q  = (const float*)d_in[1];
    const float* w_k  = (const float*)d_in[2];
    const float* w_v  = (const float*)d_in[3];
    const float* bn1s = (const float*)d_in[4];
    const float* bn1b = (const float*)d_in[5];
    const float* bn1m = (const float*)d_in[6];
    const float* bn1v = (const float*)d_in[7];
    const float* bn2s = (const float*)d_in[8];
    const float* bn2b = (const float*)d_in[9];
    const float* bn2m = (const float*)d_in[10];
    const float* bn2v = (const float*)d_in[11];
    const float* bn3s = (const float*)d_in[12];
    const float* bn3b = (const float*)d_in[13];
    const float* bn3m = (const float*)d_in[14];
    const float* bn3v = (const float*)d_in[15];
    const float* gamma = (const float*)d_in[16];
    float* out = (float*)d_out;

    char* ws = (char*)d_ws;
    unsigned short* xbt = (unsigned short*)(ws);                 // 8 MB  (b,n,c) bf16
    unsigned short* qb  = (unsigned short*)(ws +  8388608);      // 1 MB  (b,n,32) bf16
    unsigned short* kb  = (unsigned short*)(ws +  9437184);      // 1 MB  (b,n,32) bf16
    unsigned short* vb  = (unsigned short*)(ws + 10485760);      // 8 MB  (b,c,n) bf16
    unsigned short* wqb = (unsigned short*)(ws + 18874368);      // 16 KB
    unsigned short* wkb = (unsigned short*)(ws + 18890752);      // 16 KB
    unsigned short* wvb = (unsigned short*)(ws + 18907136);      // 128 KB

    k_transpose_x<<<(B_ * N_ * C_) / 256, 256, 0, stream>>>(x, xbt);
    k_convert<<<32, 256, 0, stream>>>(w_q, wqb, CR_ * C_);
    k_convert<<<32, 256, 0, stream>>>(w_k, wkb, CR_ * C_);
    k_convert<<<256, 256, 0, stream>>>(w_v, wvb, C_ * C_);

    // Q, K: 4 batches * 2 M-tiles * 256 N-tiles = 2048 wave-tiles
    k_proj<<<256, 256, 0, stream>>>(wqb, xbt, qb, bn1s, bn1b, bn1m, bn1v, CR_, 1, 2048);
    k_proj<<<256, 256, 0, stream>>>(wkb, xbt, kb, bn2s, bn2b, bn2m, bn2v, CR_, 1, 2048);
    // V: 4 * 16 * 256 = 16384 wave-tiles
    k_proj<<<2048, 256, 0, stream>>>(wvb, xbt, vb, bn3s, bn3b, bn3m, bn3v, C_, 0, 16384);

    // fused attention: 4 * 256 i-tiles, 8 waves per block
    k_attn<<<128, 256, 0, stream>>>(qb, kb, vb, x, gamma, out);
}
